// mLSTMBlock_4157528343265
// MI455X (gfx1250) — compile-verified, matrix operand order checked by reading, not run on hardware
//
#include <hip/hip_runtime.h>
#include <hip/hip_bf16.h>

// ---------------------------------------------------------------------------
// mLSTM block for MI455X (gfx1250), bf16 WMMA pipeline with async-LDS staging.
//   N=16384 tokens, D=1024, P=2048, H=1024
//   LN1 -> GEMM(up) -> GEMM(adj) -> GEMM(qkvifo fused, 6144 cols)
//       -> pointwise mLSTM + LN2 * sigmoid -> GEMM(down) + residual
// GEMMs: v_wmma_f32_16x16x32_bf16, weights pre-transposed so both A and B
// tiles stage K-contiguously via global_load_async_to_lds_b128 (ASYNCcnt),
// double-buffered in LDS.
// ---------------------------------------------------------------------------

typedef __attribute__((ext_vector_type(16))) __bf16 v16bf;
typedef __attribute__((ext_vector_type(8)))  float  v8f;

union Frag16 { v16bf v; uint u[8]; };

#if __has_builtin(__builtin_amdgcn_global_load_async_to_lds_b128)
#define ASYNC_LDS_OK 1
typedef int v4i __attribute__((vector_size(16)));           // 4 x i32, 16 bytes
typedef __attribute__((address_space(1))) v4i glb_v4i;      // global
typedef __attribute__((address_space(3))) v4i lds_v4i;      // LDS
#else
#define ASYNC_LDS_OK 0
#endif

// 16-byte memory -> LDS copy. Async (no VGPR round trip) when available.
__device__ __forceinline__ void cp_b128(const ushort* g, ushort* l) {
#if ASYNC_LDS_OK
    __builtin_amdgcn_global_load_async_to_lds_b128(
        (glb_v4i*)g, (lds_v4i*)l, /*imm offset*/0, /*cpol*/0);
#else
    *(uint4*)l = *(const uint4*)g;
#endif
}

template <int N>
__device__ __forceinline__ void wait_async() {
#if ASYNC_LDS_OK
#if __has_builtin(__builtin_amdgcn_s_wait_asynccnt)
    __builtin_amdgcn_s_wait_asynccnt(N);
#else
    asm volatile("s_wait_asynccnt %0" :: "i"(N) : "memory");
#endif
#endif
}

__device__ __forceinline__ ushort f32_to_bf16(float f) {
    uint u = __builtin_bit_cast(uint, f);
    uint r = u + 0x7FFFu + ((u >> 16) & 1u);   // round-to-nearest-even
    return (ushort)(r >> 16);
}
__device__ __forceinline__ float bf16_to_f32(ushort h) {
    uint u = ((uint)h) << 16;
    return __builtin_bit_cast(float, u);
}

// ---------------------------------------------------------------------------
// LayerNorm over last dim (1024), f32 in -> bf16 out.  One block per row.
// ---------------------------------------------------------------------------
__global__ __launch_bounds__(256)
void ln_f32_to_bf16(const float* __restrict__ x, const float* __restrict__ g,
                    const float* __restrict__ b, ushort* __restrict__ out, int D)
{
    __shared__ float red[256];
    __shared__ float red2[256];
    const int tid = threadIdx.x;
    const float* px = x + (size_t)blockIdx.x * D;

    float v[4], s = 0.f, s2 = 0.f;
    #pragma unroll
    for (int e = 0; e < 4; ++e) {
        v[e] = px[tid + e * 256];
        s += v[e]; s2 += v[e] * v[e];
    }
    red[tid] = s; red2[tid] = s2;
    __syncthreads();
    for (int off = 128; off > 0; off >>= 1) {
        if (tid < off) { red[tid] += red[tid + off]; red2[tid] += red2[tid + off]; }
        __syncthreads();
    }
    const float mu  = red[0] * (1.0f / 1024.0f);
    const float var = red2[0] * (1.0f / 1024.0f) - mu * mu;
    const float rs  = rsqrtf(var + 1e-3f);

    ushort* po = out + (size_t)blockIdx.x * D;
    #pragma unroll
    for (int e = 0; e < 4; ++e) {
        const int c = tid + e * 256;
        po[c] = f32_to_bf16((v[e] - mu) * rs * g[c] + b[c]);
    }
}

// ---------------------------------------------------------------------------
// f32 [K rows x N cols] row-major  ->  bf16 TRANSPOSED [N rows x K cols],
// written at row offset rowOff in the destination (for the 6-way qkvifo cat).
// ---------------------------------------------------------------------------
__global__ __launch_bounds__(256)
void conv_bf16_T(const float* __restrict__ src, ushort* __restrict__ dst,
                 int rowsK, int colsN, int rowOff)
{
    const int idx = blockIdx.x * 256 + threadIdx.x;
    if (idx >= rowsK * colsN) return;
    const int r = idx / colsN, c = idx % colsN;
    dst[(size_t)(rowOff + c) * rowsK + r] = f32_to_bf16(src[idx]);
}

__global__ __launch_bounds__(256)
void pack_bias6(const float* __restrict__ bq, const float* __restrict__ bk,
                const float* __restrict__ bv, const float* __restrict__ bi,
                const float* __restrict__ bf, const float* __restrict__ bo,
                float* __restrict__ dst, int H)
{
    const int idx = blockIdx.x * 256 + threadIdx.x;
    if (idx >= 6 * H) return;
    const int m = idx / H, h = idx % H;
    const float* srcs[6] = { bq, bk, bv, bi, bf, bo };
    dst[idx] = srcs[m][h];
}

// ---------------------------------------------------------------------------
// Tiled bf16 GEMM:  C[M,Ncol] = A[M,K] @ B[K,Ncol] + bias,  B given TRANSPOSED
// as Bt[Ncol,K] row-major.  Block = 256 threads = 8 waves; tile 128x128x32.
// Wave (wr,wc) in a 2x4 grid computes 4x2 WMMA 16x16 tiles.
// Double-buffered LDS, async global->LDS staging.
//   mode 0: write bf16 to Cb.   mode 1: write f32 (+residual) to Cf.
// ---------------------------------------------------------------------------
#define BM 128
#define BN 128
#define BK 32
#define LDT 40   // padded LDS row stride (ushorts): 80B, 16B-aligned, conflict-free

__global__ __launch_bounds__(256)
void gemm_bf16(const ushort* __restrict__ A, const ushort* __restrict__ Bt,
               const float* __restrict__ bias,
               ushort* __restrict__ Cb, float* __restrict__ Cf,
               const float* __restrict__ residual,
               int M, int Ncol, int K, int mode)
{
    __shared__ ushort As[2][BM * LDT];       // row-major:  As[r*LDT + k]
    __shared__ ushort Bs[2][BN * LDT];       // col-major:  Bs[c*LDT + k]

    const int tid  = threadIdx.x;
    const int lane = tid & 31;
    const int wave = tid >> 5;
    const int wr   = wave >> 2;              // 0..1
    const int wc   = wave & 3;               // 0..3
    const int row0 = blockIdx.y * BM;
    const int col0 = blockIdx.x * BN;
    const int l16  = lane & 15;
    const int half = lane >> 4;

    // per-thread staging coordinates: 512 chunks of 16B per tile, 2 per thread
    const int sr0 = tid >> 2,           sq0 = tid & 3;          // chunk 0
    const int sr1 = (tid + 256) >> 2,   sq1 = (tid + 256) & 3;  // chunk 1

    v8f acc[4][2];
    #pragma unroll
    for (int i = 0; i < 4; ++i)
        #pragma unroll
        for (int j = 0; j < 2; ++j)
            #pragma unroll
            for (int e = 0; e < 8; ++e)
                acc[i][j][e] = 0.0f;

    const ushort* gA = A  + (size_t)row0 * K;   // A rows  row0..row0+127
    const ushort* gB = Bt + (size_t)col0 * K;   // Bt rows col0..col0+127

    const int nk = K / BK;

    // prologue: stage tile 0 into buffer 0
    {
        cp_b128(gA + (size_t)sr0 * K + sq0 * 8, &As[0][sr0 * LDT + sq0 * 8]);
        cp_b128(gA + (size_t)sr1 * K + sq1 * 8, &As[0][sr1 * LDT + sq1 * 8]);
        cp_b128(gB + (size_t)sr0 * K + sq0 * 8, &Bs[0][sr0 * LDT + sq0 * 8]);
        cp_b128(gB + (size_t)sr1 * K + sq1 * 8, &Bs[0][sr1 * LDT + sq1 * 8]);
    }

    for (int kt = 0; kt < nk; ++kt) {
        const int cur = kt & 1;

        // issue async copy of the NEXT tile into the other buffer
        if (kt + 1 < nk) {
            const int k0n = (kt + 1) * BK;
            const int nxt = cur ^ 1;
            cp_b128(gA + (size_t)sr0 * K + k0n + sq0 * 8, &As[nxt][sr0 * LDT + sq0 * 8]);
            cp_b128(gA + (size_t)sr1 * K + k0n + sq1 * 8, &As[nxt][sr1 * LDT + sq1 * 8]);
            cp_b128(gB + (size_t)sr0 * K + k0n + sq0 * 8, &Bs[nxt][sr0 * LDT + sq0 * 8]);
            cp_b128(gB + (size_t)sr1 * K + k0n + sq1 * 8, &Bs[nxt][sr1 * LDT + sq1 * 8]);
            wait_async<4>();   // current tile's 4 copies done; next tile in flight
        } else {
            wait_async<0>();
        }
        __syncthreads();

        const ushort* as = As[cur];
        const ushort* bs = Bs[cur];

        // --- B fragments (32x16 each): lane = column, K-pairs per VGPR ---
        Frag16 bfr[2];
        #pragma unroll
        for (int j = 0; j < 2; ++j) {
            const int c = wc * 32 + j * 16 + l16;
            #pragma unroll
            for (int p = 0; p < 8; ++p) {
                const int kk = ((p < 4) ? (p * 2) : (16 + (p - 4) * 2)) + half * 8;
                bfr[j].u[p] = *(const uint*)(&bs[c * LDT + kk]);
            }
        }
        // --- A fragments (16x32 each) + WMMA ---
        #pragma unroll
        for (int i = 0; i < 4; ++i) {
            Frag16 afr;
            const int r = wr * 64 + i * 16 + l16;
            #pragma unroll
            for (int p = 0; p < 8; ++p) {
                const int kk = ((p < 4) ? (p * 2) : (16 + (p - 4) * 2)) + half * 8;
                afr.u[p] = *(const uint*)(&as[r * LDT + kk]);
            }
            #pragma unroll
            for (int j = 0; j < 2; ++j) {
                acc[i][j] = __builtin_amdgcn_wmma_f32_16x16x32_bf16(
                    false, afr.v, false, bfr[j].v, (short)0, acc[i][j], false, false);
            }
        }
        __syncthreads();   // all waves done reading buffer `cur` before reuse
    }

    // --- epilogue: C layout VGPR r -> M = r + 8*half, N = lane%16 ---
    #pragma unroll
    for (int i = 0; i < 4; ++i) {
        #pragma unroll
        for (int j = 0; j < 2; ++j) {
            const int n = col0 + wc * 32 + j * 16 + l16;
            const float bn = bias[n];
            #pragma unroll
            for (int r = 0; r < 8; ++r) {
                const int m = row0 + wr * 64 + i * 16 + half * 8 + r;
                const float val = acc[i][j][r] + bn;
                if (mode == 0) {
                    Cb[(size_t)m * Ncol + n] = f32_to_bf16(val);
                } else {
                    Cf[(size_t)m * Ncol + n] = val + residual[(size_t)m * Ncol + n];
                }
            }
        }
    }
}

// ---------------------------------------------------------------------------
// Pointwise mLSTM: gates -> h_t -> LN2 -> * sigmoid(x_adj).  One block/token.
// out6 row layout: [q | k | v | i | f | o], each H=1024 bf16.
// ---------------------------------------------------------------------------
__global__ __launch_bounds__(256)
void mlstm_pointwise(const ushort* __restrict__ out6, const ushort* __restrict__ xadj,
                     const float* __restrict__ g2, const float* __restrict__ b2,
                     ushort* __restrict__ u, int H)
{
    __shared__ float red[256];
    __shared__ float red2[256];
    const int tid = threadIdx.x;
    const ushort* p6 = out6 + (size_t)blockIdx.x * 6 * H;
    const ushort* pa = xadj + (size_t)blockIdx.x * H;

    float h[4], s = 0.f, s2 = 0.f;
    #pragma unroll
    for (int e = 0; e < 4; ++e) {
        const int c = tid + e * 256;
        const float q  = bf16_to_f32(p6[0 * H + c]);
        const float k  = bf16_to_f32(p6[1 * H + c]) * 0.125f;   // / sqrt(64)
        const float v  = bf16_to_f32(p6[2 * H + c]);
        const float it = bf16_to_f32(p6[3 * H + c]);
        const float ft = bf16_to_f32(p6[4 * H + c]);
        const float ot = bf16_to_f32(p6[5 * H + c]);
        const float o  = 1.0f / (1.0f + __expf(-ot));
        const float mt = fmaxf(ft, it);
        const float fg = __expf(ft - mt);
        const float ct = fg * v * k;
        const float ht = o * (ct * q) / fmaxf(fabsf(k * q), 1.0f);
        h[e] = ht;
        s += ht; s2 += ht * ht;
    }
    red[tid] = s; red2[tid] = s2;
    __syncthreads();
    for (int off = 128; off > 0; off >>= 1) {
        if (tid < off) { red[tid] += red[tid + off]; red2[tid] += red2[tid + off]; }
        __syncthreads();
    }
    const float mu  = red[0] * (1.0f / 1024.0f);
    const float var = red2[0] * (1.0f / 1024.0f) - mu * mu;
    const float rs  = rsqrtf(var + 1e-3f);

    ushort* pu = u + (size_t)blockIdx.x * H;
    #pragma unroll
    for (int e = 0; e < 4; ++e) {
        const int c = tid + e * 256;
        const float xa = bf16_to_f32(pa[c]);
        const float sg = 1.0f / (1.0f + __expf(-xa));
        pu[c] = f32_to_bf16(((h[e] - mu) * rs * g2[c] + b2[c]) * sg);
    }
}

// ---------------------------------------------------------------------------
// Host-side orchestration
// ---------------------------------------------------------------------------
extern "C" void kernel_launch(void* const* d_in, const int* in_sizes, int n_in,
                              void* d_out, int out_size, void* d_ws, size_t ws_size,
                              hipStream_t stream)
{
    (void)in_sizes; (void)n_in; (void)out_size; (void)ws_size;
    const int Ntok = 8 * 2048;     // 16384
    const int D = 1024, P = 2048, H = 1024;

    const float* x     = (const float*)d_in[0];
    const float* ln1_g = (const float*)d_in[1];
    const float* ln1_b = (const float*)d_in[2];
    const float* W_up  = (const float*)d_in[3];
    const float* b_up  = (const float*)d_in[4];
    const float* W_adj = (const float*)d_in[5];
    const float* b_adj = (const float*)d_in[6];
    const float* W_q   = (const float*)d_in[7];
    const float* b_q   = (const float*)d_in[8];
    const float* W_k   = (const float*)d_in[9];
    const float* b_k   = (const float*)d_in[10];
    const float* W_v   = (const float*)d_in[11];
    const float* b_v   = (const float*)d_in[12];
    const float* W_i   = (const float*)d_in[13];
    const float* b_i   = (const float*)d_in[14];
    const float* W_f   = (const float*)d_in[15];
    const float* b_f   = (const float*)d_in[16];
    const float* W_o   = (const float*)d_in[17];
    const float* b_o   = (const float*)d_in[18];
    const float* ln2_g = (const float*)d_in[19];
    const float* ln2_b = (const float*)d_in[20];
    const float* W_dn  = (const float*)d_in[21];
    const float* b_dn  = (const float*)d_in[22];
    float* out = (float*)d_out;

    // workspace layout (bytes)
    char* ws = (char*)d_ws;
    size_t off = 0;
    auto take = [&](size_t bytes) { char* p = ws + off; off += (bytes + 255) & ~(size_t)255; return p; };
    ushort* xn    = (ushort*)take((size_t)Ntok * D * 2);       // LN1 output (bf16)
    ushort* xup   = (ushort*)take((size_t)Ntok * P * 2);       // up-proj (bf16)
    ushort* xadj  = (ushort*)take((size_t)Ntok * H * 2);       // adj-proj (bf16)
    ushort* out6  = (ushort*)take((size_t)Ntok * 6 * H * 2);   // q|k|v|i|f|o (bf16)
    ushort* ubuf  = (ushort*)take((size_t)Ntok * H * 2);       // LN2*sig (bf16)
    ushort* wupT  = (ushort*)take((size_t)P * D * 2);          // W_up^T  [2048 x 1024]
    ushort* wadjT = (ushort*)take((size_t)H * P * 2);          // W_adj^T [1024 x 2048]
    ushort* wcatT = (ushort*)take((size_t)6 * H * H * 2);      // Wcat^T  [6144 x 1024]
    ushort* wdnT  = (ushort*)take((size_t)D * H * 2);          // W_down^T[1024 x 1024]
    float*  bcat  = (float*)take((size_t)6 * H * 4);

    // 1) LN1
    ln_f32_to_bf16<<<Ntok, 256, 0, stream>>>(x, ln1_g, ln1_b, xn, D);

    // 2) weight conversions (transposed bf16)
    conv_bf16_T<<<(D * P + 255) / 256, 256, 0, stream>>>(W_up,  wupT,  D, P, 0);
    conv_bf16_T<<<(P * H + 255) / 256, 256, 0, stream>>>(W_adj, wadjT, P, H, 0);
    const float* wg[6] = { W_q, W_k, W_v, W_i, W_f, W_o };
    for (int m = 0; m < 6; ++m)
        conv_bf16_T<<<(H * H + 255) / 256, 256, 0, stream>>>(wg[m], wcatT, H, H, m * H);
    conv_bf16_T<<<(H * D + 255) / 256, 256, 0, stream>>>(W_dn, wdnT, H, D, 0);
    pack_bias6<<<(6 * H + 255) / 256, 256, 0, stream>>>(b_q, b_k, b_v, b_i, b_f, b_o, bcat, H);

    // 3) GEMM chain
    dim3 blk(256);
    // x_up = xn @ W_up + b_up            [16384 x 2048]
    gemm_bf16<<<dim3(P / BN, Ntok / BM), blk, 0, stream>>>(
        xn, wupT, b_up, xup, nullptr, nullptr, Ntok, P, D, 0);
    // x_adj = x_up @ W_adj + b_adj       [16384 x 1024]
    gemm_bf16<<<dim3(H / BN, Ntok / BM), blk, 0, stream>>>(
        xup, wadjT, b_adj, xadj, nullptr, nullptr, Ntok, H, P, 0);
    // out6 = x_adj @ Wcat + bcat         [16384 x 6144]
    gemm_bf16<<<dim3(6 * H / BN, Ntok / BM), blk, 0, stream>>>(
        xadj, wcatT, bcat, out6, nullptr, nullptr, Ntok, 6 * H, H, 0);

    // 4) pointwise mLSTM + LN2 + sigmoid gate
    mlstm_pointwise<<<Ntok, 256, 0, stream>>>(out6, xadj, ln2_g, ln2_b, ubuf, H);

    // 5) out = u @ W_down + b_down + x   [16384 x 1024], f32
    gemm_bf16<<<dim3(D / BN, Ntok / BM), blk, 0, stream>>>(
        ubuf, wdnT, b_dn, nullptr, out, x, Ntok, D, H, 1);
}
